// SwitchTransformersBlock_29858612642050
// MI455X (gfx1250) — compile-verified, matrix operand order checked by reading, not run on hardware
//
#include <hip/hip_runtime.h>

// Problem constants (match reference)
#define BB   8
#define SS   2048
#define DD   1024
#define FF   4096
#define EE   8
#define TT   (BB * SS)        // 16384 tokens
#define EPSF 1e-6f

#define TILE_M   32           // tokens per FFN block (2 row-tiles of 16)
#define TILES_PE (TT / TILE_M)// 512 tiles per expert (upper bound)
#define LDX      (DD + 8)     // padded LDS stride for X tile (bank-conflict free b128)
#define FCHUNK   128          // F processed per outer iteration
#define LDH      (FCHUNK + 8) // padded LDS stride for h chunk

// Dynamic LDS layout for the FFN kernel
#define XS_ELEMS (TILE_M * LDX)                 // 33,024 bf16 = 66,048 B
#define HS_ELEMS (TILE_M * LDH)                 //  4,352 bf16 =  8,704 B
#define SMEM_BYTES (XS_ELEMS * 2 + HS_ELEMS * 2 + TILE_M * 4)   // ~74.9 KB (<320 KB WGP LDS)

typedef __attribute__((ext_vector_type(16))) __bf16 v16bf;
typedef __attribute__((ext_vector_type(8)))  float  v8f;

union FragB {
  v16bf v;
  uint4 q[2];
};

__device__ __forceinline__ unsigned short f2bf(float f) {
  unsigned u = __float_as_uint(f);
  u += 0x7fffu + ((u >> 16) & 1u);      // round-to-nearest-even
  return (unsigned short)(u >> 16);
}

// A fragment (16x32 bf16, MxK) from LDS, row-major with stride ld.
// ISA layout: lane m=lane&15, h=lane>>4; elems 0..7 = K 8h+0..7, elems 8..15 = K 16+8h+0..7.
__device__ __forceinline__ v16bf a_frag_lds(const unsigned short* m0, int ld, int k0, int lane) {
  int m = lane & 15, h = lane >> 4;
  const unsigned short* p = m0 + m * ld + k0 + 8 * h;
  FragB f;
  f.q[0] = *(const uint4*)p;
  f.q[1] = *(const uint4*)(p + 16);
  return f.v;
}

// B fragment (32x16 bf16, KxN) from pre-transposed global weights WT[n][k] (K contiguous).
// lane n=lane&15, h=lane>>4; same K-per-element pattern as A => two b128 loads.
__device__ __forceinline__ v16bf bt_frag_g(const unsigned short* WT, int ldk, int row, int k0, int lane) {
  int n = lane & 15, h = lane >> 4;
  const unsigned short* p = WT + (size_t)(row + n) * ldk + k0 + 8 * h;
  FragB f;
  f.q[0] = *(const uint4*)p;
  f.q[1] = *(const uint4*)(p + 16);
  return f.v;
}

// ---------------- kernel 0: zero per-expert counters ----------------
__global__ void zero_counts_kernel(int* __restrict__ counts) {
  if (threadIdx.x < EE) counts[threadIdx.x] = 0;
}

// ---------------- kernel 1: fp32 -> bf16 transposing conversion ----------------
// src: [E][R][C] f32 ; dst: [E][C][R] bf16.  grid = (C/32, R/32, E), 256 threads.
__global__ __launch_bounds__(256)
void cvt_bf16_t_kernel(const float* __restrict__ src, unsigned short* __restrict__ dst,
                       int R, int C) {
  __shared__ unsigned short T[32][34];   // pad -> 17-bank stride, conflict-free
  const int tid = threadIdx.x;
  const int tx = tid & 31, ty = tid >> 5;          // 32 x 8
  const int r0 = blockIdx.y * 32, c0 = blockIdx.x * 32;
  const float* s = src + (size_t)blockIdx.z * R * C;
  unsigned short* d = dst + (size_t)blockIdx.z * R * C;

#pragma unroll
  for (int p = 0; p < 4; p++) {
    int r = ty + 8 * p;
    T[r][tx] = f2bf(s[(size_t)(r0 + r) * C + c0 + tx]);
  }
  __syncthreads();
#pragma unroll
  for (int p = 0; p < 4; p++) {
    int cc = ty + 8 * p;
    d[(size_t)(c0 + cc) * R + r0 + tx] = T[tx][cc];
  }
}

// ---------------- kernel 2: RMSNorm + top-1 router ----------------
__global__ __launch_bounds__(256)
void norm_router_kernel(const float* __restrict__ hs, const float* __restrict__ lnw,
                        const float* __restrict__ rw,
                        unsigned short* __restrict__ normed,
                        float* __restrict__ logits_out, float* __restrict__ expidx_out,
                        float* __restrict__ topprob,
                        int* __restrict__ counts, int* __restrict__ tlist) {
  const int token = blockIdx.x;
  const int tid = threadIdx.x, lane = tid & 31, wv = tid >> 5;
  const float* x = hs + (size_t)token * DD;

  float xv[4];
  float ss = 0.f;
#pragma unroll
  for (int i = 0; i < 4; i++) {
    xv[i] = x[tid + 256 * i];
    ss += xv[i] * xv[i];
  }
#pragma unroll
  for (int o = 16; o > 0; o >>= 1) ss += __shfl_xor(ss, o, 32);

  __shared__ float red[8];
  __shared__ float lred[EE];
  if (lane == 0) red[wv] = ss;
  if (tid < EE) lred[tid] = 0.f;
  __syncthreads();

  float tot = 0.f;
#pragma unroll
  for (int i = 0; i < 8; i++) tot += red[i];
  const float rstd = rsqrtf(tot * (1.0f / DD) + EPSF);

  float lg[EE];
#pragma unroll
  for (int e = 0; e < EE; e++) lg[e] = 0.f;
#pragma unroll
  for (int i = 0; i < 4; i++) {
    int d = tid + 256 * i;
    float nv = xv[i] * rstd * lnw[d];
    normed[(size_t)token * DD + d] = f2bf(nv);
    const float* r = rw + (size_t)d * EE;
#pragma unroll
    for (int e = 0; e < EE; e++) lg[e] += nv * r[e];
  }
#pragma unroll
  for (int e = 0; e < EE; e++) {
    float v = lg[e];
#pragma unroll
    for (int o = 16; o > 0; o >>= 1) v += __shfl_xor(v, o, 32);
    if (lane == 0) atomicAdd(&lred[e], v);
  }
  __syncthreads();

  if (tid == 0) {
    float l[EE];
    float mx = -1e30f;
    int am = 0;
#pragma unroll
    for (int e = 0; e < EE; e++) {
      l[e] = lred[e];
      if (l[e] > mx) { mx = l[e]; am = e; }   // strict '>' => first max (jnp.argmax tie rule)
    }
    float s = 0.f;
#pragma unroll
    for (int e = 0; e < EE; e++) s += __expf(l[e] - mx);
    float tp = 1.0f / s;                      // softmax at the argmax: exp(0)/sum
#pragma unroll
    for (int e = 0; e < EE; e++) logits_out[(size_t)token * EE + e] = l[e];
    expidx_out[token] = (float)am;
    topprob[token]   = tp;
    int slot = atomicAdd(&counts[am], 1);
    tlist[am * TT + slot] = token;
  }
}

// ---------------- kernel 3: per-expert fused FFN (gather -> GEMM -> ReLU -> GEMM -> combine) ----------------
// 32-token tile, 8 waves; wave w owns h cols 16w..16w+15 (GEMM1) and out cols 128w..128w+127 (GEMM2).
__global__ __launch_bounds__(256)
void expert_ffn_kernel(const float* __restrict__ hs,
                       const unsigned short* __restrict__ normed,
                       const unsigned short* __restrict__ wiT,   // [E][F][D] bf16 (K=d contiguous)
                       const unsigned short* __restrict__ woT,   // [E][D][F] bf16 (K=f contiguous)
                       const float* __restrict__ topprob,
                       const int* __restrict__ counts, const int* __restrict__ tlist,
                       float* __restrict__ out) {
  const int e    = blockIdx.x >> 9;           // / TILES_PE (=512)
  const int tile = blockIdx.x & (TILES_PE - 1);
  const int cnt  = counts[e];
  if (tile * TILE_M >= cnt) return;

  extern __shared__ char smem[];
  unsigned short* Xs  = (unsigned short*)smem;                 // [TILE_M][LDX]
  unsigned short* Hsm = Xs + XS_ELEMS;                         // [TILE_M][LDH]
  int*            toks = (int*)(Hsm + HS_ELEMS);               // [TILE_M]

  const int tid = threadIdx.x, lane = tid & 31, w = tid >> 5;
  const int n0 = lane & 15, h = lane >> 4;

  if (tid < TILE_M) {
    int idx = tile * TILE_M + tid;
    toks[tid] = (idx < cnt) ? tlist[e * TT + idx] : -1;
  }
  __syncthreads();

  // Cooperative load of the 32x1024 bf16 activation tile into LDS (zero-pad short tiles).
  {
    int r = tid >> 3;                 // 32 rows, 8 threads each
    int cb = (tid & 7) * 128;         // 128 bf16 per thread
    int tok = toks[r];
    if (tok >= 0) {
      const unsigned short* src = normed + (size_t)tok * DD + cb;
#pragma unroll
      for (int i = 0; i < 16; i++)
        *(uint4*)&Xs[r * LDX + cb + 8 * i] = *(const uint4*)(src + 8 * i);
    } else {
      uint4 z = {0u, 0u, 0u, 0u};
#pragma unroll
      for (int i = 0; i < 16; i++)
        *(uint4*)&Xs[r * LDX + cb + 8 * i] = z;
    }
  }
  __syncthreads();

  const unsigned short* WiT = wiT + (size_t)e * DD * FF;  // row f, K=d, ldk=DD
  const unsigned short* WoT = woT + (size_t)e * FF * DD;  // row d, K=f, ldk=FF

  v8f accD[2][8];
#pragma unroll
  for (int u = 0; u < 2; u++)
#pragma unroll
    for (int t = 0; t < 8; t++)
#pragma unroll
      for (int j = 0; j < 8; j++) accD[u][t][j] = 0.f;

#pragma unroll 1
  for (int fc = 0; fc < FF; fc += FCHUNK) {
    // --- GEMM 1: h[32 x 128] = X[32 x 1024] * Wi[:, fc:fc+128]
    v8f acch[2];
#pragma unroll
    for (int u = 0; u < 2; u++)
#pragma unroll
      for (int j = 0; j < 8; j++) acch[u][j] = 0.f;
#pragma unroll 4
    for (int k0 = 0; k0 < DD; k0 += 32) {
      v16bf b  = bt_frag_g(WiT, DD, fc + 16 * w, k0, lane);
      v16bf a0 = a_frag_lds(Xs, LDX, k0, lane);
      v16bf a1 = a_frag_lds(Xs + 16 * LDX, LDX, k0, lane);
      acch[0] = __builtin_amdgcn_wmma_f32_16x16x32_bf16(false, a0, false, b, (short)0, acch[0], false, false);
      acch[1] = __builtin_amdgcn_wmma_f32_16x16x32_bf16(false, a1, false, b, (short)0, acch[1], false, false);
    }
    // ReLU -> bf16 -> LDS
#pragma unroll
    for (int u = 0; u < 2; u++)
#pragma unroll
      for (int j = 0; j < 8; j++)
        Hsm[(u * 16 + h * 8 + j) * LDH + 16 * w + n0] = f2bf(fmaxf(acch[u][j], 0.f));
    __syncthreads();

    // --- GEMM 2: out[32 x 1024] += h[32 x 128] * Wo[fc:fc+128, :]
#pragma unroll 2
    for (int k0 = 0; k0 < FCHUNK; k0 += 32) {
      v16bf a0 = a_frag_lds(Hsm, LDH, k0, lane);
      v16bf a1 = a_frag_lds(Hsm + 16 * LDH, LDH, k0, lane);
#pragma unroll
      for (int t = 0; t < 8; t++) {
        v16bf b = bt_frag_g(WoT, FF, 128 * w + 16 * t, fc + k0, lane);
        accD[0][t] = __builtin_amdgcn_wmma_f32_16x16x32_bf16(false, a0, false, b, (short)0, accD[0][t], false, false);
        accD[1][t] = __builtin_amdgcn_wmma_f32_16x16x32_bf16(false, a1, false, b, (short)0, accD[1][t], false, false);
      }
    }
    __syncthreads();
  }

  // --- Epilogue: out = hidden + top_prob * acc, scattered back per token.
#pragma unroll
  for (int u = 0; u < 2; u++) {
#pragma unroll
    for (int j = 0; j < 8; j++) {
      int r = u * 16 + h * 8 + j;
      int tok = toks[r];
      if (tok >= 0) {
        float tp = topprob[tok];
        const float* hrow = hs + (size_t)tok * DD;
        float* orow = out + (size_t)tok * DD;
#pragma unroll
        for (int t = 0; t < 8; t++) {
          int c = 128 * w + 16 * t + n0;
          orow[c] = hrow[c] + tp * accD[u][t][j];
        }
      }
    }
  }
}

extern "C" void kernel_launch(void* const* d_in, const int* in_sizes, int n_in,
                              void* d_out, int out_size, void* d_ws, size_t ws_size,
                              hipStream_t stream) {
  const float* hs  = (const float*)d_in[0];   // [B,S,D]
  const float* lnw = (const float*)d_in[1];   // [D]
  const float* rw  = (const float*)d_in[2];   // [D,E]
  const float* wi  = (const float*)d_in[3];   // [E,D,F]
  const float* wo  = (const float*)d_in[4];   // [E,F,D]

  float* out        = (float*)d_out;                          // [B,S,D]
  float* logits_out = out + (size_t)TT * DD;                  // [B,S,E]
  float* expidx_out = logits_out + (size_t)TT * EE;           // [B,S]

  // Workspace carve-up (~168.5 MB total, 256B-aligned sections).
  char* p = (char*)d_ws;
  auto carve = [&](size_t bytes) {
    void* q = (void*)p;
    p += (bytes + 255) & ~(size_t)255;
    return q;
  };
  unsigned short* normed = (unsigned short*)carve((size_t)TT * DD * 2);          // 32 MB
  unsigned short* wibT   = (unsigned short*)carve((size_t)EE * DD * FF * 2);     // 64 MB
  unsigned short* wobT   = (unsigned short*)carve((size_t)EE * FF * DD * 2);     // 64 MB
  float*          tprob  = (float*)carve((size_t)TT * 4);                        // 64 KB
  int*            counts = (int*)carve((size_t)EE * 4);
  int*            tlist  = (int*)carve((size_t)EE * TT * 4);                     // 512 KB
  (void)ws_size; (void)in_sizes; (void)n_in; (void)out_size;

  // 0) reset per-expert counters (ws is stale between replays)
  zero_counts_kernel<<<1, 32, 0, stream>>>(counts);

  // 1) weights -> transposed bf16 (K-contiguous for B fragments)
  //    wi [E][D][F] -> wibT [E][F][D] ; wo [E][F][D] -> wobT [E][D][F]
  cvt_bf16_t_kernel<<<dim3(FF / 32, DD / 32, EE), 256, 0, stream>>>(wi, wibT, DD, FF);
  cvt_bf16_t_kernel<<<dim3(DD / 32, FF / 32, EE), 256, 0, stream>>>(wo, wobT, FF, DD);

  // 2) RMSNorm + router + token gather lists
  norm_router_kernel<<<TT, 256, 0, stream>>>(hs, lnw, rw, normed, logits_out,
                                             expidx_out, tprob, counts, tlist);

  // 3) per-expert WMMA FFN + weighted combine + residual (dynamic LDS ~75 KB)
  expert_ffn_kernel<<<EE * TILES_PE, 256, SMEM_BYTES, stream>>>(hs, normed, wibT, wobT, tprob,
                                                                counts, tlist, out);
}
